// RegionProposalNetwork_1967095022272
// MI455X (gfx1250) — compile-verified, matrix operand order checked by reading, not run on hardware
//
#include <hip/hip_runtime.h>
#include <math.h>

// ---------------------------------------------------------------------------
// RPN pipeline for MI455X (gfx1250, wave32, WMMA, async-to-LDS):
//   1) convert_weights : w_sw fp32 -> bf16 row-major [2048][18432]
//   2) im2col_pack     : input fp32 -> bf16 B-fragments in WMMA lane layout
//   3) conv3x3_wmma    : implicit GEMM; 16 waves/block (4x4), 2x2 tiles/wave,
//                        double-buffered LDS staged with
//                        global_load_async_to_lds_b128 + s_wait_asynccnt
//   4) heads           : 1x1 convs (reg 36ch, cls 18ch) fp32 VALU -> d_out
//   5) decode          : anchor decode -> boxes, sigmoid scores, sort init
//   6) bitonic_step    : global-memory bitonic argsort (descending)
//   7) nms_kernel      : greedy NMS, 1 workgroup, 221KB dynamic LDS
// ---------------------------------------------------------------------------

typedef __attribute__((ext_vector_type(16))) __bf16    v16bf;
typedef __attribute__((ext_vector_type(8)))  float     v8f;
typedef __attribute__((ext_vector_type(4)))  unsigned  u32x4;

#define NBOX   9216
#define NPAD   16384
#define KDIM   18432     // 2048*9
#define KT     576       // KDIM/32
#define KCH    288       // KT/2 (2 kt-steps per LDS chunk)
#define COUT   2048
#define NPIX   1024      // 32*32

__device__ __forceinline__ unsigned short f2bf(float f) {
    unsigned u = __float_as_uint(f);
    u = (u + 0x7FFFu + ((u >> 16) & 1u)) >> 16;   // round-to-nearest-even
    return (unsigned short)u;
}

__device__ __forceinline__ v8f wmma_bf16(v16bf a, v16bf b, v8f c) {
    return __builtin_amdgcn_wmma_f32_16x16x32_bf16(
        false, a, false, b, (short)0, c, false, false);
}

// async 16B copy global -> LDS (CDNA5; tracked by ASYNCcnt)
__device__ __forceinline__ void async_cp16(const void* gaddr, unsigned ldsaddr) {
    asm volatile("global_load_async_to_lds_b128 %0, %1, off"
                 :: "v"(ldsaddr), "v"(gaddr) : "memory");
}

// ---- 1) fp32 -> bf16 weight conversion -------------------------------------
__global__ void convert_weights(const float* __restrict__ w,
                                unsigned short* __restrict__ wbf, int n) {
    int stride = gridDim.x * blockDim.x;
    for (int i = blockIdx.x * blockDim.x + threadIdx.x; i < n; i += stride)
        wbf[i] = f2bf(w[i]);
}

// ---- 2) im2col into WMMA B-fragment layout ---------------------------------
// Fragment (kt, pt): lane l holds B[K = kt*32 + kk(l,i), pix = pt*16 + (l&15)]
// kk(l,i) = (l<16 ? 0 : 8) + (i<8 ? i : i+8)   (mirror of 16-bit A layout)
__global__ void im2col_pack(const float* __restrict__ in,
                            unsigned short* __restrict__ bpack) {
    int t = blockIdx.x * blockDim.x + threadIdx.x;
    if (t >= KT * 64 * 32) return;
    int lane = t & 31;
    int pt   = (t >> 5) & 63;
    int kt   = t >> 11;
    int pix  = pt * 16 + (lane & 15);
    int y = pix >> 5, x = pix & 31;
    int base = (lane < 16) ? 0 : 8;
    unsigned short out[16];
#pragma unroll
    for (int i = 0; i < 16; ++i) {
        int kk = base + (i < 8 ? i : i + 8);
        int k  = kt * 32 + kk;
        int cin = k / 9, r = k - cin * 9;
        int dy = r / 3, dx = r - dy * 3;
        int iy = y + dy - 1, ix = x + dx - 1;
        float v = 0.0f;
        if (iy >= 0 && iy < 32 && ix >= 0 && ix < 32)
            v = in[(size_t)cin * NPIX + iy * 32 + ix];
        out[i] = f2bf(v);
    }
    unsigned short* dst = bpack + (size_t)kt * 32768 + (size_t)pt * 512 + lane * 16;
    *(u32x4*)dst       = *(const u32x4*)out;
    *(u32x4*)(dst + 8) = *(const u32x4*)(out + 8);
}

// ---- 3) main conv: blocked implicit GEMM via WMMA + async LDS staging ------
// Block: 512 thr = 16 waves (4x4). Block tile: 8 ct-tiles x 8 pt-tiles.
// Wave (wy,wx) computes 2x2 16x16 tiles: ct {2wy,2wy+1} x pt {2wx,2wx+1}.
// LDS: double-buffered chunk of 2 kt-steps: lA/lB [buf][ktc][tile][512 bf16].
__global__ __launch_bounds__(512)
void conv3x3_wmma(const unsigned short* __restrict__ wbf,
                  const unsigned short* __restrict__ bpack,
                  const float* __restrict__ bias,
                  float* __restrict__ sw) {
    __shared__ unsigned short lA[2][2][8][512];   // 32 KB
    __shared__ unsigned short lB[2][2][8][512];   // 32 KB

    const int tid  = threadIdx.x;
    const int lane = tid & 31;
    const int wave = tid >> 5;
    const int wy = wave >> 2, wx = wave & 3;
    const int half = (lane < 16) ? 0 : 8;          // K-half select (A/B layout)

    const int ctrow0 = blockIdx.y * 128;           // first cout row of block
    const int ptg0   = blockIdx.x * 8;             // first pt-tile of block

    // stage one chunk (2 kt-steps): 4 async b128 copies per thread
    auto stage = [&](int kc, int bufi) {
#pragma unroll
        for (int r = 0; r < 2; ++r) {
            int ta   = tid + r * 512;              // 0..1023
            int frag = ta >> 6;                    // 0..15
            int ktc  = frag >> 3, tile = frag & 7;
            int e    = ta & 63;
            int ln   = e >> 1, ch = e & 1;
            int kb   = (kc * 2 + ktc) * 32;
            // A: weights row-major -> fragment layout
            {
                int row = ctrow0 + tile * 16 + (ln & 15);
                int col = kb + ((ln < 16) ? 0 : 8) + ch * 16;
                const void* g = wbf + (size_t)row * KDIM + col;
                unsigned l = (unsigned)(size_t)(const void*)
                             &lA[bufi][ktc][tile][ln * 16 + ch * 8];
                async_cp16(g, l);
            }
            // B: pre-packed fragments, contiguous copy
            {
                const void* g = bpack + (size_t)(kc * 2 + ktc) * 32768
                              + (size_t)(ptg0 + tile) * 512 + ln * 16 + ch * 8;
                unsigned l = (unsigned)(size_t)(const void*)
                             &lB[bufi][ktc][tile][ln * 16 + ch * 8];
                async_cp16(g, l);
            }
        }
    };

    v8f acc00 = {0,0,0,0,0,0,0,0}, acc01 = {0,0,0,0,0,0,0,0};
    v8f acc10 = {0,0,0,0,0,0,0,0}, acc11 = {0,0,0,0,0,0,0,0};

    stage(0, 0);
    int buf = 0;
    for (int kc = 0; kc < KCH; ++kc) {
        if (kc + 1 < KCH) {
            stage(kc + 1, buf ^ 1);                    // prefetch next chunk
            asm volatile("s_wait_asynccnt 4" ::: "memory");  // chunk kc landed
        } else {
            asm volatile("s_wait_asynccnt 0" ::: "memory");
        }
        __syncthreads();
#pragma unroll
        for (int ktc = 0; ktc < 2; ++ktc) {
            v16bf a0 = *(const v16bf*)&lA[buf][ktc][2 * wy + 0][lane * 16];
            v16bf a1 = *(const v16bf*)&lA[buf][ktc][2 * wy + 1][lane * 16];
            v16bf b0 = *(const v16bf*)&lB[buf][ktc][2 * wx + 0][lane * 16];
            v16bf b1 = *(const v16bf*)&lB[buf][ktc][2 * wx + 1][lane * 16];
            acc00 = wmma_bf16(a0, b0, acc00);
            acc01 = wmma_bf16(a0, b1, acc01);
            acc10 = wmma_bf16(a1, b0, acc10);
            acc11 = wmma_bf16(a1, b1, acc11);
        }
        __syncthreads();                                // LDS free for reuse
        buf ^= 1;
    }

    // C/D layout: lanes 0-15 -> M = v, lanes 16-31 -> M = v+8; N = lane&15
#pragma unroll
    for (int i = 0; i < 2; ++i) {
#pragma unroll
        for (int j = 0; j < 2; ++j) {
            v8f acc = (i == 0) ? (j == 0 ? acc00 : acc01)
                               : (j == 0 ? acc10 : acc11);
            int ct = (ctrow0 >> 4) + 2 * wy + i;
            int pix = (ptg0 + 2 * wx + j) * 16 + (lane & 15);
#pragma unroll
            for (int v = 0; v < 8; ++v) {
                int cout = ct * 16 + v + half;
                sw[(size_t)cout * NPIX + pix] = acc[v] + bias[cout];
            }
        }
    }
}

// ---- 4) 1x1 conv heads (reg: 36 ch, cls: 18 ch), direct to d_out -----------
__global__ void heads(const float* __restrict__ sw,
                      const float* __restrict__ wreg, const float* __restrict__ breg,
                      const float* __restrict__ wcls, const float* __restrict__ bcls,
                      float* __restrict__ out) {
    int t = blockIdx.x * blockDim.x + threadIdx.x;     // 55296 threads
    if (t < 36 * NPIX) {
        int co = t / NPIX, pix = t - co * NPIX;
        const float* wr = wreg + (size_t)co * COUT;
        float acc = breg[co];
        for (int c = 0; c < COUT; ++c) acc += wr[c] * sw[(size_t)c * NPIX + pix];
        int kk = co >> 2, ch = co & 3;
        out[36864 + (pix * 9 + kk) * 4 + ch] = acc;    // reg region
    } else if (t < 54 * NPIX) {
        int t2 = t - 36 * NPIX;
        int co = t2 / NPIX, pix = t2 - co * NPIX;
        const float* wc = wcls + (size_t)co * COUT;
        float acc = bcls[co];
        for (int c = 0; c < COUT; ++c) acc += wc[c] * sw[(size_t)c * NPIX + pix];
        int kk = co >> 1, ch = co & 1;
        out[73728 + (pix * 9 + kk) * 2 + ch] = acc;    // cls region
    }
}

// ---- 5) box decode + objectness score + sort-buffer init -------------------
__global__ void decode(const float* __restrict__ anchors,
                       float* __restrict__ out,
                       float* __restrict__ scores, int* __restrict__ sidx) {
    int n = blockIdx.x * blockDim.x + threadIdx.x;     // 16384 threads
    if (n >= NPAD) return;
    if (n < NBOX) {
        float ax = anchors[n*4+0], ay = anchors[n*4+1];
        float aw = anchors[n*4+2], ah = anchors[n*4+3];
        float acx = ax + aw * 0.5f, acy = ay + ah * 0.5f;
        const float* rg = out + 36864 + n * 4;
        float cx = acx + aw * rg[0];
        float cy = acy + ah * rg[1];
        float w  = aw * __expf(rg[2]);
        float h  = ah * __expf(rg[3]);
        out[n*4+0] = cx - w * 0.5f;
        out[n*4+1] = cy - h * 0.5f;
        out[n*4+2] = cx + w * 0.5f;
        out[n*4+3] = cy + h * 0.5f;
        float c0 = out[73728 + n*2], c1 = out[73728 + n*2 + 1];
        scores[n] = 1.0f / (1.0f + __expf(c0 - c1));   // softmax[:,1]
        sidx[n] = n;
    } else {
        scores[n] = -__builtin_huge_valf();
        sidx[n] = n;
    }
}

// ---- 6) bitonic argsort step (descending; stable tie-break on index) -------
__global__ void bitonic_step(float* __restrict__ s, int* __restrict__ id,
                             int j, int k) {
    int i = blockIdx.x * blockDim.x + threadIdx.x;
    int ixj = i ^ j;
    if (ixj <= i || ixj >= NPAD) return;
    float si = s[i], sj = s[ixj];
    int   ii = id[i], ij = id[ixj];
    bool jBetter = (sj > si) || (sj == si && ij < ii);
    if ((((i & k) == 0)) == jBetter) {
        s[i] = sj; s[ixj] = si;
        id[i] = ij; id[ixj] = ii;
    }
}

// ---- 7) greedy NMS: 1 workgroup, all boxes in LDS (needs CDNA5 320KB) ------
__global__ __launch_bounds__(1024)
void nms_kernel(const float* __restrict__ boxes,   // d_out base (xyxy)
                const int* __restrict__ order,
                float* __restrict__ keep_out) {
    extern __shared__ float smem[];
    float* X1 = smem;
    float* Y1 = X1 + NBOX;
    float* X2 = Y1 + NBOX;
    float* Y2 = X2 + NBOX;
    float* AR = Y2 + NBOX;
    int*  SUP = (int*)(AR + NBOX);

    for (int n = threadIdx.x; n < NBOX; n += 1024) {
        int o = order[n];
        float x1 = boxes[o*4+0], y1 = boxes[o*4+1];
        float x2 = boxes[o*4+2], y2 = boxes[o*4+3];
        X1[n] = x1; Y1[n] = y1; X2[n] = x2; Y2[n] = y2;
        AR[n] = (x2 - x1) * (y2 - y1);
        SUP[n] = 0;
    }
    __syncthreads();

    for (int i = 0; i < NBOX - 1; ++i) {
        if (SUP[i] == 0) {
            float bx1 = X1[i], by1 = Y1[i], bx2 = X2[i], by2 = Y2[i], ba = AR[i];
            for (int n = i + 1 + threadIdx.x; n < NBOX; n += 1024) {
                float ix1 = fmaxf(bx1, X1[n]);
                float iy1 = fmaxf(by1, Y1[n]);
                float ix2 = fminf(bx2, X2[n]);
                float iy2 = fminf(by2, Y2[n]);
                float inter = fmaxf(ix2 - ix1, 0.f) * fmaxf(iy2 - iy1, 0.f);
                float iou = inter / (ba + AR[n] - inter + 1e-9f);
                if (iou > 0.7f) SUP[n] = 1;
            }
        }
        __syncthreads();
    }

    for (int n = threadIdx.x; n < NBOX; n += 1024)
        keep_out[order[n]] = SUP[n] ? 0.0f : 1.0f;
}

// ---------------------------------------------------------------------------
extern "C" void kernel_launch(void* const* d_in, const int* in_sizes, int n_in,
                              void* d_out, int out_size, void* d_ws, size_t ws_size,
                              hipStream_t stream) {
    const float* in_fm   = (const float*)d_in[0];   // (1,2048,32,32)
    const float* w_sw    = (const float*)d_in[1];   // (2048,2048,3,3)
    const float* b_sw    = (const float*)d_in[2];
    const float* w_reg   = (const float*)d_in[3];
    const float* b_reg   = (const float*)d_in[4];
    const float* w_cls   = (const float*)d_in[5];
    const float* b_cls   = (const float*)d_in[6];
    const float* anchors = (const float*)d_in[7];
    float* out = (float*)d_out;   // boxes[0..36863] reg[36864..] cls[73728..] keep[92160..]

    // workspace carve-up (~122 MB total)
    char* ws = (char*)d_ws;
    unsigned short* wbf   = (unsigned short*)(ws);                 // 75,497,472 B
    unsigned short* bpack = (unsigned short*)(ws + 75497472);      // 37,748,736 B
    float*          sw    = (float*)(ws + 113246208);              //  8,388,608 B
    float*          scores= (float*)(ws + 121634816);              //     65,536 B
    int*            sidx  = (int*)  (ws + 121700352);              //     65,536 B

    convert_weights<<<4096, 256, 0, stream>>>(w_sw, wbf, COUT * KDIM);
    im2col_pack<<<(KT * 64 * 32 + 255) / 256, 256, 0, stream>>>(in_fm, bpack);

    dim3 cgrid(8, 16);                                 // 8 pt-blocks x 16 ct-blocks
    conv3x3_wmma<<<cgrid, 512, 0, stream>>>(wbf, bpack, b_sw, sw);

    heads<<<(54 * NPIX + 255) / 256, 256, 0, stream>>>(sw, w_reg, b_reg, w_cls, b_cls, out);
    decode<<<NPAD / 256, 256, 0, stream>>>(anchors, out, scores, sidx);

    for (int k = 2; k <= NPAD; k <<= 1)
        for (int j = k >> 1; j > 0; j >>= 1)
            bitonic_step<<<NPAD / 256, 256, 0, stream>>>(scores, sidx, j, k);

    size_t smem = (size_t)NBOX * 5 * sizeof(float) + (size_t)NBOX * sizeof(int);
    nms_kernel<<<1, 1024, smem, stream>>>(out, sidx, out + 92160);
}
